// GPT2Model_56727928045888
// MI455X (gfx1250) — matrix operand order repaired and checked by reading, not compiled, not run on hardware
//
#include <hip/hip_runtime.h>

// ---------------- problem dims ----------------
constexpr int kB  = 2;
constexpr int kS  = 1024;
constexpr int kV  = 50257;
constexpr int kD  = 768;
constexpr int kH  = 12;
constexpr int kL  = 6;
constexpr int kF  = 3072;
constexpr int kDK = 64;

typedef __attribute__((ext_vector_type(16))) __bf16        v16bf;
typedef __attribute__((ext_vector_type(8)))  float         v8f;
typedef __attribute__((ext_vector_type(4)))  unsigned int  u32x4;

// Load a 16-element bf16 WMMA fragment as two aligned 16B chunks at p[0..7]
// and p[gap..gap+7]. gap=16 for A-layout fragments and for
// B-layout fragments.
__device__ __forceinline__ v16bf load_frag(const __bf16* p, int gap) {
  union { u32x4 q[2]; v16bf v; } u;
  u.q[0] = *reinterpret_cast<const u32x4*>(p);
  u.q[1] = *reinterpret_cast<const u32x4*>(p + gap);
  return u.v;
}

// Async global -> LDS copy of 16 bytes per lane (ASYNCcnt-tracked).
// lds byte address = low 32 bits of the flat LDS pointer (ISA 10.2 aperture).
__device__ __forceinline__ void async_copy_b128(void* lds_ptr, const void* gptr) {
  unsigned lds_addr = (unsigned)(unsigned long long)lds_ptr;
  unsigned long long ga = (unsigned long long)gptr;
  asm volatile("global_load_async_to_lds_b128 %0, %1, off"
               :: "v"(lds_addr), "v"(ga) : "memory");
}
__device__ __forceinline__ void wait_async0() {
  asm volatile("s_wait_asynccnt 0" ::: "memory");
}

// ---------------- fp32 -> bf16 weight conversion ----------------
__global__ void convert_kernel(const float* __restrict__ src,
                               __bf16* __restrict__ dst, long n) {
  long i = (long)blockIdx.x * blockDim.x + threadIdx.x;
  long stride = (long)gridDim.x * blockDim.x;
  for (; i < n; i += stride) dst[i] = (__bf16)src[i];
}

// ---------------- embedding ----------------
__global__ __launch_bounds__(256) void embed_kernel(
    const int* __restrict__ ids, const float* __restrict__ tok,
    const float* __restrict__ pos, float* __restrict__ x) {
  const int row = blockIdx.x;           // 0 .. B*S-1
  const int s   = row % kS;
  const int id  = ids[row];
  const float* t  = tok + (size_t)id * kD;
  const float* pp = pos + (size_t)s * kD;
  float* xr = x + (size_t)row * kD;
  for (int i = threadIdx.x; i < kD; i += 256) xr[i] = t[i] + pp[i];
}

// ---------------- layernorm (fp32 in -> bf16 out) ----------------
__global__ __launch_bounds__(256) void layernorm_kernel(
    const float* __restrict__ x, const float* __restrict__ g,
    const float* __restrict__ b, __bf16* __restrict__ out) {
  const int row = blockIdx.x;
  const float* xr = x + (size_t)row * kD;
  float s = 0.f, s2 = 0.f;
  for (int i = threadIdx.x; i < kD; i += 256) {
    float v = xr[i]; s += v; s2 += v * v;
  }
  for (int o = 16; o > 0; o >>= 1) {
    s  += __shfl_xor(s,  o);
    s2 += __shfl_xor(s2, o);
  }
  __shared__ float rs[8], rs2[8];
  const int w = threadIdx.x >> 5, lane = threadIdx.x & 31;
  if (lane == 0) { rs[w] = s; rs2[w] = s2; }
  __syncthreads();
  if (threadIdx.x == 0) {
    float a = 0.f, a2 = 0.f;
    for (int i = 0; i < 8; ++i) { a += rs[i]; a2 += rs2[i]; }
    rs[0] = a; rs2[0] = a2;
  }
  __syncthreads();
  const float mu   = rs[0] / (float)kD;
  const float var  = rs2[0] / (float)kD - mu * mu;
  const float rstd = rsqrtf(var + 1e-5f);
  __bf16* outr = out + (size_t)row * kD;
  for (int i = threadIdx.x; i < kD; i += 256)
    outr[i] = (__bf16)((xr[i] - mu) * rstd * g[i] + b[i]);
}

// ---------------- WMMA bf16 GEMM ----------------
// C[M,N] = A[M,K](bf16, row-major) @ B[K,N](bf16, row-major), f32 accumulate.
// Block tile 128x128, 8 waves, each wave a 32x64 tile (2x4 accumulators).
// A tile staged with async global->LDS copies; B tile staged transposed.
// mode 0: outF = acc                     (fp32 out, no bias)
// mode 1: resid += acc + bias            (fp32 residual update in place)
// mode 2: outH = bf16(acc)               (no bias)
// mode 3: outH = bf16(gelu(acc + bias))  (exact gelu)
constexpr int BM = 128, BN = 128, BK = 32, AP = 40;  // AP: padded LDS pitch

__global__ __launch_bounds__(256) void gemm_bf16_kernel(
    const __bf16* __restrict__ A, const __bf16* __restrict__ Bm,
    const float* __restrict__ bias, float* __restrict__ resid,
    float* __restrict__ outF, __bf16* __restrict__ outH,
    int Mdim, int Ndim, int Kdim, int mode) {
  __shared__ __align__(16) __bf16 As[BM * AP];  // [m][k]
  __shared__ __align__(16) __bf16 Bs[BN * AP];  // transposed: [n][k]

  const int tid  = threadIdx.x;
  const int lane = tid & 31, w = tid >> 5;
  const int half = lane >> 4, l15 = lane & 15;
  const int wm = (w & 3) * 32;     // wave tile offsets inside block tile
  const int wn = (w >> 2) * 64;
  const int m0 = blockIdx.y * BM;
  const int n0 = blockIdx.x * BN;

  v8f acc[2][4] = {};
  const bool nvec = (Ndim % 8 == 0);

  for (int k0 = 0; k0 < Kdim; k0 += BK) {
    // stage A tile asynchronously: 128 rows x 32 bf16 = 512 b128 chunks
    for (int c = tid; c < BM * 4; c += 256) {
      const int row = c >> 2, cc = (c & 3) * 8;
      async_copy_b128(&As[row * AP + cc],
                      A + (size_t)(m0 + row) * Kdim + k0 + cc);
    }
    // stage B tile transposed: 32 k-rows x 128 n-cols in 8-col chunks
    for (int c = tid; c < 512; c += 256) {
      const int kk = c >> 4;
      const int nn = (c & 15) * 8;
      const __bf16* src = Bm + (size_t)(k0 + kk) * Ndim + n0 + nn;
      __bf16 tmp[8];
      if (nvec && (n0 + nn + 8 <= Ndim)) {
        *reinterpret_cast<u32x4*>(tmp) = *reinterpret_cast<const u32x4*>(src);
      } else {
        for (int e = 0; e < 8; ++e)
          tmp[e] = (n0 + nn + e < Ndim) ? src[e] : (__bf16)0.f;
      }
      for (int e = 0; e < 8; ++e) Bs[(nn + e) * AP + kk] = tmp[e];
    }
    wait_async0();
    __syncthreads();

    v16bf af[2], bfr[4];
    for (int i = 0; i < 2; ++i)
      af[i] = load_frag(&As[(wm + i * 16 + l15) * AP + 8 * half], 16);
    for (int j = 0; j < 4; ++j)
      bfr[j] = load_frag(&Bs[(wn + j * 16 + l15) * AP + 8 * half], 16);
    for (int i = 0; i < 2; ++i)
      for (int j = 0; j < 4; ++j)
        acc[i][j] = __builtin_amdgcn_wmma_f32_16x16x32_bf16(
            false, af[i], false, bfr[j], (short)0, acc[i][j], false, false);
    __syncthreads();
  }

  // epilogue: C/D layout -> lane holds col n = l15, rows m = r + 8*half
  for (int i = 0; i < 2; ++i)
    for (int j = 0; j < 4; ++j) {
      const int n = n0 + wn + j * 16 + l15;
      if (n >= Ndim) continue;
      const float bv = (bias != nullptr) ? bias[n] : 0.f;
      for (int r = 0; r < 8; ++r) {
        const int m = m0 + wm + i * 16 + 8 * half + r;
        const size_t off = (size_t)m * Ndim + n;
        const float vv = acc[i][j][r];
        if (mode == 0) {
          outF[off] = vv;
        } else if (mode == 1) {
          resid[off] += vv + bv;
        } else if (mode == 2) {
          outH[off] = (__bf16)vv;
        } else {
          const float t = vv + bv;
          outH[off] = (__bf16)(0.5f * t * (1.f + erff(t * 0.70710678118f)));
        }
      }
    }
}

// ---------------- flash attention ----------------
// 4 waves per block; block covers 64 consecutive queries (wave w -> 16 of
// them). K tiles staged with async global->LDS copies; V staged transposed.
// q,k,v,ctx are bf16 [B, S, D] with head h occupying cols h*64 .. h*64+63.
constexpr int KP = 72;  // Ks LDS pitch (dk row), bytes%16==0
constexpr int VP = 40;  // Vt LDS pitch (key row)

__global__ __launch_bounds__(128) void attention_kernel(
    const __bf16* __restrict__ q, const __bf16* __restrict__ k,
    const __bf16* __restrict__ v, __bf16* __restrict__ ctx) {
  __shared__ __align__(16) __bf16 Ks[32 * KP];      // [key_local][dk]
  __shared__ __align__(16) __bf16 Vt[64 * VP];      // [dk][key_local]
  __shared__ __align__(16) __bf16 Pl[4][16 * 32];   // per-wave [q][key_local]

  const int tid  = threadIdx.x;
  const int lane = tid & 31, w = tid >> 5;
  const int half = lane >> 4, l15 = lane & 15;
  const int i0_blk = blockIdx.x * 64;
  const int i0 = i0_blk + w * 16;      // this wave's query tile
  const int h = blockIdx.y, b = blockIdx.z;
  const size_t base = (size_t)b * kS * kD + (size_t)h * kDK;
  const __bf16* qp = q + base;
  const __bf16* kp = k + base;
  const __bf16* vp = v + base;
  __bf16* op = ctx + base;
  __bf16* pl = &Pl[w][0];

  // Q A-fragments for dk 0..31 and 32..63 (held for the whole loop)
  v16bf qa[2];
  for (int f = 0; f < 2; ++f)
    qa[f] = load_frag(qp + (size_t)(i0 + l15) * kD + 32 * f + 8 * half, 16);

  v8f o[4] = {};
  float run_max[8], run_sum[8];
  for (int r = 0; r < 8; ++r) { run_max[r] = -3.0e38f; run_sum[r] = 0.f; }

  const int nt = (i0_blk + 64 + 31) >> 5;  // uniform trip count per block
  for (int t = 0; t < nt; ++t) {
    const int j0 = t * 32;
    // ---- cooperative staging: K tile async, V tile transposed ----
    for (int c = tid; c < 256; c += 128) {
      const int row = c >> 3;          // key_local 0..31
      const int cc  = (c & 7) * 8;     // dk chunk
      async_copy_b128(&Ks[row * KP + cc], kp + (size_t)(j0 + row) * kD + cc);
      __bf16 tmp[8];
      *reinterpret_cast<u32x4*>(tmp) =
          *reinterpret_cast<const u32x4*>(vp + (size_t)(j0 + row) * kD + cc);
      for (int e = 0; e < 8; ++e) Vt[(cc + e) * VP + row] = tmp[e];
    }
    wait_async0();
    __syncthreads();

    if (j0 < i0 + 16) {  // wave-uniform causal bound
      // S tile = Q @ K^T for two 16-key sub-tiles
      v8f sc[2];
      for (int ks = 0; ks < 2; ++ks) {
        const int keyl = 16 * ks + l15;
        const v16bf kb0 = load_frag(&Ks[keyl * KP + 0  + 8 * half], 16);
        const v16bf kb1 = load_frag(&Ks[keyl * KP + 32 + 8 * half], 16);
        v8f z = {};
        z = __builtin_amdgcn_wmma_f32_16x16x32_bf16(false, qa[0], false, kb0,
                                                    (short)0, z, false, false);
        z = __builtin_amdgcn_wmma_f32_16x16x32_bf16(false, qa[1], false, kb1,
                                                    (short)0, z, false, false);
        sc[ks] = z;
      }
      // scale, causal mask, online softmax (row stats per m = r + 8*half)
      float p[2][8];
      for (int r = 0; r < 8; ++r) {
        const int qi = i0 + 8 * half + r;
        float mx = -3.0e38f;
        for (int ks = 0; ks < 2; ++ks) {
          const int key = j0 + 16 * ks + l15;
          float sv = sc[ks][r] * 0.125f;  // 1/sqrt(64)
          if (key > qi) sv = -1e9f;
          sc[ks][r] = sv;
          mx = fmaxf(mx, sv);
        }
        for (int d = 1; d < 16; d <<= 1) mx = fmaxf(mx, __shfl_xor(mx, d));
        const float nm   = fmaxf(run_max[r], mx);
        const float corr = __expf(run_max[r] - nm);
        run_max[r] = nm;
        float rsum = 0.f;
        for (int ks = 0; ks < 2; ++ks) {
          const float pv = __expf(sc[ks][r] - nm);
          p[ks][r] = pv;
          rsum += pv;
        }
        for (int d = 1; d < 16; d <<= 1) rsum += __shfl_xor(rsum, d);
        run_sum[r] = run_sum[r] * corr + rsum;
        for (int jn = 0; jn < 4; ++jn) o[jn][r] *= corr;
      }
      // re-lay-out P through this wave's LDS slab into A-fragment form
      for (int ks = 0; ks < 2; ++ks)
        for (int r = 0; r < 8; ++r)
          pl[(8 * half + r) * 32 + 16 * ks + l15] = (__bf16)p[ks][r];
      asm volatile("s_wait_dscnt 0" ::: "memory");
      const v16bf pa = load_frag(&pl[l15 * 32 + 8 * half], 16);
      // O += P @ V (4 output column tiles of dk) from transposed V in LDS
      for (int jn = 0; jn < 4; ++jn) {
        const v16bf vb = load_frag(&Vt[(jn * 16 + l15) * VP + 8 * half], 16);
        o[jn] = __builtin_amdgcn_wmma_f32_16x16x32_bf16(
            false, pa, false, vb, (short)0, o[jn], false, false);
      }
      asm volatile("s_wait_dscnt 0" ::: "memory");
    }
    __syncthreads();
  }
  // normalize and write ctx (bf16)
  for (int jn = 0; jn < 4; ++jn)
    for (int r = 0; r < 8; ++r) {
      const float val = o[jn][r] / run_sum[r];
      op[(size_t)(i0 + 8 * half + r) * kD + jn * 16 + l15] = (__bf16)val;
    }
}

// ---------------- launch ----------------
extern "C" void kernel_launch(void* const* d_in, const int* in_sizes, int n_in,
                              void* d_out, int out_size, void* d_ws, size_t ws_size,
                              hipStream_t stream) {
  (void)in_sizes; (void)n_in; (void)out_size; (void)ws_size;
  const int*   ids   = (const int*)d_in[0];
  const float* tok   = (const float*)d_in[1];
  const float* pos   = (const float*)d_in[2];
  const float* wq    = (const float*)d_in[3];
  const float* wk    = (const float*)d_in[4];
  const float* wv    = (const float*)d_in[5];
  const float* wo    = (const float*)d_in[6];
  const float* bo    = (const float*)d_in[7];
  const float* fcw   = (const float*)d_in[8];
  const float* fcb   = (const float*)d_in[9];
  const float* prw   = (const float*)d_in[10];
  const float* prb   = (const float*)d_in[11];
  const float* ln1g  = (const float*)d_in[12];
  const float* ln1b  = (const float*)d_in[13];
  const float* ln2g  = (const float*)d_in[14];
  const float* ln2b  = (const float*)d_in[15];
  const float* lnfg  = (const float*)d_in[16];
  const float* lnfb  = (const float*)d_in[17];
  const float* outw  = (const float*)d_in[18];

  // carve workspace
  char* p = (char*)d_ws;
  auto take = [&](size_t bytes) -> char* {
    char* r = p;
    p += (bytes + 255) & ~(size_t)255;
    return r;
  };
  const size_t ndd = (size_t)kL * kD * kD;
  const size_t ndf = (size_t)kL * kD * kF;
  __bf16* wq_h  = (__bf16*)take(ndd * 2);
  __bf16* wk_h  = (__bf16*)take(ndd * 2);
  __bf16* wv_h  = (__bf16*)take(ndd * 2);
  __bf16* wo_h  = (__bf16*)take(ndd * 2);
  __bf16* fcw_h = (__bf16*)take(ndf * 2);
  __bf16* prw_h = (__bf16*)take(ndf * 2);
  __bf16* ow_h  = (__bf16*)take((size_t)kD * kV * 2);
  float*  x     = (float*)take((size_t)kB * kS * kD * 4);
  __bf16* xn    = (__bf16*)take((size_t)kB * kS * kD * 2);
  __bf16* qb    = (__bf16*)take((size_t)kB * kS * kD * 2);
  __bf16* kb    = (__bf16*)take((size_t)kB * kS * kD * 2);
  __bf16* vb    = (__bf16*)take((size_t)kB * kS * kD * 2);
  __bf16* cx    = (__bf16*)take((size_t)kB * kS * kD * 2);
  __bf16* hb    = (__bf16*)take((size_t)kB * kS * kF * 2);

  const int M = kB * kS;  // 2048

  // weights -> bf16 (every call; deterministic)
  convert_kernel<<<1024, 256, 0, stream>>>(wq,  wq_h,  (long)ndd);
  convert_kernel<<<1024, 256, 0, stream>>>(wk,  wk_h,  (long)ndd);
  convert_kernel<<<1024, 256, 0, stream>>>(wv,  wv_h,  (long)ndd);
  convert_kernel<<<1024, 256, 0, stream>>>(wo,  wo_h,  (long)ndd);
  convert_kernel<<<1024, 256, 0, stream>>>(fcw, fcw_h, (long)ndf);
  convert_kernel<<<1024, 256, 0, stream>>>(prw, prw_h, (long)ndf);
  convert_kernel<<<1024, 256, 0, stream>>>(outw, ow_h, (long)kD * kV);

  embed_kernel<<<M, 256, 0, stream>>>(ids, tok, pos, x);

  const dim3 gD(kD / BN, M / BM);    // N=768 -> 6 x 16
  const dim3 gF(kF / BN, M / BM);    // N=3072 -> 24 x 16
  for (int l = 0; l < kL; ++l) {
    layernorm_kernel<<<M, 256, 0, stream>>>(x, ln1g + l * kD, ln1b + l * kD, xn);
    gemm_bf16_kernel<<<gD, 256, 0, stream>>>(xn, wq_h + (size_t)l * kD * kD,
        nullptr, nullptr, nullptr, qb, M, kD, kD, 2);
    gemm_bf16_kernel<<<gD, 256, 0, stream>>>(xn, wk_h + (size_t)l * kD * kD,
        nullptr, nullptr, nullptr, kb, M, kD, kD, 2);
    gemm_bf16_kernel<<<gD, 256, 0, stream>>>(xn, wv_h + (size_t)l * kD * kD,
        nullptr, nullptr, nullptr, vb, M, kD, kD, 2);
    attention_kernel<<<dim3(kS / 64, kH, kB), 128, 0, stream>>>(qb, kb, vb, cx);
    gemm_bf16_kernel<<<gD, 256, 0, stream>>>(cx, wo_h + (size_t)l * kD * kD,
        bo + l * kD, x, nullptr, nullptr, M, kD, kD, 1);
    layernorm_kernel<<<M, 256, 0, stream>>>(x, ln2g + l * kD, ln2b + l * kD, xn);
    gemm_bf16_kernel<<<gF, 256, 0, stream>>>(xn, fcw_h + (size_t)l * kD * kF,
        fcb + l * kF, nullptr, nullptr, hb, M, kF, kD, 3);
    gemm_bf16_kernel<<<gD, 256, 0, stream>>>(hb, prw_h + (size_t)l * kF * kD,
        prb + l * kD, x, nullptr, nullptr, M, kD, kF, 1);
  }
  layernorm_kernel<<<M, 256, 0, stream>>>(x, lnfg, lnfb, xn);
  const dim3 gV((kV + BN - 1) / BN, M / BM);
  gemm_bf16_kernel<<<gV, 256, 0, stream>>>(xn, ow_h, nullptr, nullptr,
      (float*)d_out, nullptr, M, kV, kD, 0);
}
